// HybridGenerator_62835371540598
// MI455X (gfx1250) — matrix operand order repaired and checked by reading, not compiled, not run on hardware
//
#include <hip/hip_runtime.h>
#include <cmath>

#define N_NODES 20000
#define HID 512
#define NHEAD 4
#define CHN 128
#define LRELU_SLOPE 0.2f
#define EPS_LN 1e-5f

typedef __attribute__((ext_vector_type(16))) __bf16        v16bf;
typedef __attribute__((ext_vector_type(8)))  float         v8f;
typedef __attribute__((ext_vector_type(4)))  unsigned int  u32x4;

union BFrag { u32x4 q[2]; v16bf v; };
union CFrag { v8f v; float f[8]; };

__device__ __forceinline__ unsigned short f2bf(float f) {
  unsigned int u = __float_as_uint(f);
  u += 0x7FFFu + ((u >> 16) & 1u);          // round to nearest even
  return (unsigned short)(u >> 16);
}

// ---------------- generic helpers ----------------
__global__ void k_fill(float* p, float v, int n) {
  int i = blockIdx.x * blockDim.x + threadIdx.x;
  if (i < n) p[i] = v;
}

__global__ void k_f32_to_bf16(const float* s, unsigned short* d, int n) {
  int i = blockIdx.x * blockDim.x + threadIdx.x;
  if (i < n) d[i] = f2bf(s[i]);
}

// W[K][Nc] f32 -> BT[Nc][K] bf16 (transposed so B-fragment loads are contiguous)
__global__ void k_wt_to_bf16_t(const float* w, unsigned short* bt, int K, int Nc) {
  int i = blockIdx.x * blockDim.x + threadIdx.x;
  if (i >= K * Nc) return;
  int k = i / Nc, n = i % Nc;
  bt[(size_t)n * K + k] = f2bf(w[i]);
}

// ---------------- WMMA GEMM: C[M,Nc] = A[M,K] @ B[K,Nc] (+bias)(+relu) ----------------
// A: bf16 row-major [M][K]; BT: bf16 [Nc][K] (B transposed).
// Register blocking: each wave owns a 32x64 C macro-tile = 2 M-tiles x 4 N-tiles,
// i.e. 8 v_wmma_f32_16x16x32_bf16 per k-step against 12 b128 loads (~21 FLOP/B).
// M must be a multiple of 32 (20000 = 625*32); Nc a multiple of 64.
__global__ void k_wmma_gemm(const unsigned short* __restrict__ A,
                            const unsigned short* __restrict__ BT,
                            const float* __restrict__ bias, float* __restrict__ C,
                            int M, int K, int Nc, int act) {
  const int lane = threadIdx.x & 31;
  const int wave = threadIdx.x >> 5;
  const int tmp  = blockIdx.x * 8 + wave;   // index over 32-row M pairs
  const int pairsM = M >> 5;
  if (tmp >= pairsM) return;                // uniform per wave: EXEC stays all-1s
  const int tn0  = blockIdx.y * 4;          // first of 4 consecutive 16-col N tiles
  const int half = lane >> 4;
  const int l15  = lane & 15;

  const int m0 = (tmp << 5) + l15;          // A rows owned by this lane
  const unsigned short* arow0 = A + (size_t)m0 * K;
  const unsigned short* arow1 = arow0 + (size_t)16 * K;
  const unsigned short* brow[4];
#pragma unroll
  for (int j = 0; j < 4; ++j)
    brow[j] = BT + (size_t)((tn0 + j) * 16 + l15) * K;

  CFrag acc[2][4];
#pragma unroll
  for (int i = 0; i < 2; ++i)
#pragma unroll
    for (int j = 0; j < 4; ++j)
#pragma unroll
      for (int r = 0; r < 8; ++r) acc[i][j].f[r] = 0.0f;

  for (int k0 = 0; k0 < K; k0 += 32) {
    __builtin_prefetch(arow0 + k0 + 32, 0, 1);    // global_prefetch_b8
    __builtin_prefetch(arow1 + k0 + 32, 0, 1);
    BFrag fa0, fa1, fb[4];
    // 16-bit A 16x32 layout: lanes 0-15 hold K 0..7 & 16..23; lanes 16-31 hold K 8..15 & 24..31
    fa0.q[0] = *reinterpret_cast<const u32x4*>(arow0 + k0 + half * 8);
    fa0.q[1] = *reinterpret_cast<const u32x4*>(arow0 + k0 + 16 + half * 8);
    fa1.q[0] = *reinterpret_cast<const u32x4*>(arow1 + k0 + half * 8);
    fa1.q[1] = *reinterpret_cast<const u32x4*>(arow1 + k0 + 16 + half * 8);
    // 16-bit B 32x16 layout: lanes 0-15 hold K 0..7 & 16..23 of col n; lanes 16-31 hold K 8..15 & 24..31
#pragma unroll
    for (int j = 0; j < 4; ++j) {
      fb[j].q[0] = *reinterpret_cast<const u32x4*>(brow[j] + k0 + half * 8);
      fb[j].q[1] = *reinterpret_cast<const u32x4*>(brow[j] + k0 + 16 + half * 8);
    }
#pragma unroll
    for (int j = 0; j < 4; ++j) {
      acc[0][j].v = __builtin_amdgcn_wmma_f32_16x16x32_bf16(false, fa0.v, false, fb[j].v,
                                                            (short)0, acc[0][j].v,
                                                            false, false);
      acc[1][j].v = __builtin_amdgcn_wmma_f32_16x16x32_bf16(false, fa1.v, false, fb[j].v,
                                                            (short)0, acc[1][j].v,
                                                            false, false);
    }
  }
#pragma unroll
  for (int j = 0; j < 4; ++j) {
    const int n = (tn0 + j) * 16 + l15;
    const float bv = bias ? bias[n] : 0.0f;
#pragma unroll
    for (int i = 0; i < 2; ++i) {
#pragma unroll
      for (int r = 0; r < 8; ++r) {         // C/D layout: VGPR r -> row r (lanes 0-15), r+8 (16-31)
        int row = (tmp << 5) + i * 16 + r + half * 8;
        float v = acc[i][j].f[r] + bv;
        if (act == 1) v = v > 0.f ? v : 0.f;
        C[(size_t)row * Nc + n] = v;
      }
    }
  }
}

// ---------------- embed: h = x[N,4] @ W[4,128] + b ----------------
__global__ void k_embed(const float* x, const float* W, const float* b, float* h, int n) {
  int i = blockIdx.x * blockDim.x + threadIdx.x;
  if (i >= n * 128) return;
  int node = i >> 7, j = i & 127;
  float acc = b[j];
#pragma unroll
  for (int t = 0; t < 4; ++t) acc += x[node * 4 + t] * W[t * 128 + j];
  h[i] = acc;
}

// ---------------- GAT per-node attention scores ----------------
__global__ void k_scores(const float* xl, const float* asrc, const float* adst,
                         float* ss, float* sd, int n) {
  int i = blockIdx.x * blockDim.x + threadIdx.x;
  if (i >= n * NHEAD) return;
  int node = i >> 2, hd = i & 3;
  const float* xr = xl + (size_t)node * HID + hd * CHN;
  const float* as = asrc + hd * CHN;
  const float* ad = adst + hd * CHN;
  float a = 0.f, d = 0.f;
  for (int c = 0; c < CHN; ++c) { float v = xr[c]; a += v * as[c]; d += v * ad[c]; }
  ss[i] = a; sd[i] = d;
}

__device__ __forceinline__ void atomicMaxF(float* addr, float val) {
  if (val >= 0.f) atomicMax((int*)addr, __float_as_int(val));
  else            atomicMin((unsigned int*)addr, __float_as_uint(val));
}

__device__ __forceinline__ void edge_decode(const long long* ei, int e, int Eraw,
                                            int& s, int& d) {
  if (e < Eraw) { s = (int)ei[e]; d = (int)ei[Eraw + e]; }
  else          { s = e - Eraw;   d = s; }                 // self loops
}

__global__ void k_edge_max(const long long* ei, const float* ss, const float* sd,
                           float* mm, int Eraw, int E) {
  int e = blockIdx.x * blockDim.x + threadIdx.x;
  if (e >= E) return;
  int s, d; edge_decode(ei, e, Eraw, s, d);
#pragma unroll
  for (int h = 0; h < 4; ++h) {
    float l = ss[s * 4 + h] + sd[d * 4 + h];
    l = l > 0.f ? l : LRELU_SLOPE * l;
    atomicMaxF(&mm[d * 4 + h], l);
  }
}

__global__ void k_edge_den(const long long* ei, const float* ss, const float* sd,
                           const float* mm, float* den, int Eraw, int E) {
  int e = blockIdx.x * blockDim.x + threadIdx.x;
  if (e >= E) return;
  int s, d; edge_decode(ei, e, Eraw, s, d);
#pragma unroll
  for (int h = 0; h < 4; ++h) {
    float l = ss[s * 4 + h] + sd[d * 4 + h];
    l = l > 0.f ? l : LRELU_SLOPE * l;
    atomicAdd(&den[d * 4 + h], __expf(l - mm[d * 4 + h]));
  }
}

// one wave per edge: out[dst] += alpha * xl[src] across all 512 channels
__global__ void k_edge_agg(const long long* ei, const float* ss, const float* sd,
                           const float* mm, const float* den, const float* xl,
                           float* out, int Eraw, int E) {
  int lane = threadIdx.x & 31;
  int e = blockIdx.x * 8 + (threadIdx.x >> 5);
  if (e >= E) return;
  int s, d; edge_decode(ei, e, Eraw, s, d);
  float alpha[4];
#pragma unroll
  for (int h = 0; h < 4; ++h) {
    float l = ss[s * 4 + h] + sd[d * 4 + h];
    l = l > 0.f ? l : LRELU_SLOPE * l;
    alpha[h] = __expf(l - mm[d * 4 + h]) / den[d * 4 + h];
  }
  const float* xr = xl + (size_t)s * HID;
  float* orow = out + (size_t)d * HID;
#pragma unroll
  for (int i = 0; i < 16; ++i) {
    int ch = i * 32 + lane;
    atomicAdd(&orow[ch], alpha[ch >> 7] * xr[ch]);
  }
}

// ---------------- +bias, LayerNorm(512), ELU — one wave per node ----------------
__global__ void k_bias_ln_elu(float* hio, const float* b, const float* g,
                              const float* beta, int n) {
  int lane = threadIdx.x & 31;
  int node = blockIdx.x * 8 + (threadIdx.x >> 5);
  if (node >= n) return;
  float* row = hio + (size_t)node * HID;
  float v[16];
  float s = 0.f;
#pragma unroll
  for (int i = 0; i < 16; ++i) {
    int ch = i * 32 + lane;
    v[i] = row[ch] + b[ch];
    s += v[i];
  }
#pragma unroll
  for (int off = 16; off; off >>= 1) s += __shfl_xor(s, off, 32);
  float mu = s * (1.f / 512.f);
  float q = 0.f;
#pragma unroll
  for (int i = 0; i < 16; ++i) { float t = v[i] - mu; q += t * t; }
#pragma unroll
  for (int off = 16; off; off >>= 1) q += __shfl_xor(q, off, 32);
  float rv = rsqrtf(q * (1.f / 512.f) + EPS_LN);
#pragma unroll
  for (int i = 0; i < 16; ++i) {
    int ch = i * 32 + lane;
    float y = (v[i] - mu) * rv * g[ch] + beta[ch];
    row[ch] = y > 0.f ? y : __expf(y) - 1.f;   // ELU
  }
}

// ---------------- CNN (single image, NCHW), conv 4x4 s2 p1 ----------------
__global__ void k_conv(const float* in, const float* w, const float* b, float* out,
                       int Ci, int Hi, int Co, int Ho) {
  int i = blockIdx.x * blockDim.x + threadIdx.x;
  int tot = Co * Ho * Ho;
  if (i >= tot) return;
  int o = i / (Ho * Ho), r = i % (Ho * Ho);
  int oy = r / Ho, ox = r % Ho;
  float acc = b[o];
  for (int ci = 0; ci < Ci; ++ci) {
    const float* ip = in + (size_t)ci * Hi * Hi;
    const float* wp = w + ((size_t)o * Ci + ci) * 16;
#pragma unroll
    for (int ky = 0; ky < 4; ++ky) {
      int iy = oy * 2 - 1 + ky;
      if ((unsigned)iy >= (unsigned)Hi) continue;
#pragma unroll
      for (int kx = 0; kx < 4; ++kx) {
        int ix = ox * 2 - 1 + kx;
        if ((unsigned)ix >= (unsigned)Hi) continue;
        acc += ip[iy * Hi + ix] * wp[ky * 4 + kx];
      }
    }
  }
  out[i] = acc;
}

__global__ void k_bn_stats(const float* y, float* mu, float* rv, int HW) {
  __shared__ float ssum[256], ssq[256];
  int c = blockIdx.x;
  const float* p = y + (size_t)c * HW;
  float s = 0.f, q = 0.f;
  for (int i = threadIdx.x; i < HW; i += 256) { float v = p[i]; s += v; q += v * v; }
  ssum[threadIdx.x] = s; ssq[threadIdx.x] = q;
  __syncthreads();
  for (int off = 128; off; off >>= 1) {
    if ((int)threadIdx.x < off) {
      ssum[threadIdx.x] += ssum[threadIdx.x + off];
      ssq[threadIdx.x]  += ssq[threadIdx.x + off];
    }
    __syncthreads();
  }
  if (threadIdx.x == 0) {
    float m = ssum[0] / HW;
    float var = ssq[0] / HW - m * m;
    mu[c] = m; rv[c] = rsqrtf(var + EPS_LN);
  }
}

__global__ void k_bn_lrelu(float* y, const float* mu, const float* rv,
                           const float* g, const float* beta, int HW, int tot) {
  int i = blockIdx.x * blockDim.x + threadIdx.x;
  if (i >= tot) return;
  int c = i / HW;
  float v = (y[i] - mu[c]) * rv[c] * g[c] + beta[c];
  y[i] = v > 0.f ? v : LRELU_SLOPE * v;
}

// proj: y[256] = flat[8192] @ W[8192,256] + b ; one wave per output
__global__ void k_proj(const float* f, const float* W, const float* b, float* y) {
  int lane = threadIdx.x & 31;
  int o = blockIdx.x * 8 + (threadIdx.x >> 5);
  if (o >= 256) return;
  float s = 0.f;
  for (int i = lane; i < 8192; i += 32) s += f[i] * W[(size_t)i * 256 + o];
#pragma unroll
  for (int off = 16; off; off >>= 1) s += __shfl_xor(s, off, 32);
  if (lane == 0) y[o] = s + b[o];
}

// k = y@Wk+bk, v = y@Wv+bv (each [128]); kv[0:128]=k, kv[128:256]=v
__global__ void k_kv(const float* y, const float* Wk, const float* bk,
                     const float* Wv, const float* bv, float* kv) {
  int t = blockIdx.x * blockDim.x + threadIdx.x;
  if (t >= 256) return;
  const float *W, *bb; float* outp; int j;
  if (t < 128) { W = Wk; bb = bk; j = t;       outp = kv; }
  else         { W = Wv; bb = bv; j = t - 128; outp = kv + 128; }
  float s = bb[j];
  for (int i = 0; i < 256; ++i) s += y[i] * W[i * 128 + j];
  outp[j] = s;
}

// softmax over a length-1 axis == 1, so attention output is q + v broadcast
__global__ void k_addv(float* q, const float* v, int n) {
  int i = blockIdx.x * blockDim.x + threadIdx.x;
  if (i >= n * 128) return;
  q[i] += v[i & 127];
}

// final 512 -> 4 + sigmoid
__global__ void k_dec2(const float* h, const float* W, const float* b, float* out, int n) {
  int i = blockIdx.x * blockDim.x + threadIdx.x;
  if (i >= n * 4) return;
  int node = i >> 2, o = i & 3;
  const float* hr = h + (size_t)node * HID;
  float s = b[o];
  for (int c = 0; c < HID; ++c) s += hr[c] * W[c * 4 + o];
  out[i] = 1.f / (1.f + __expf(-s));
}

// ======================================================================
extern "C" void kernel_launch(void* const* d_in, const int* in_sizes, int n_in,
                              void* d_out, int out_size, void* d_ws, size_t ws_size,
                              hipStream_t stream) {
  const int N = N_NODES;
  auto cdiv = [](int a, int b) { return (a + b - 1) / b; };

  // ---- inputs (setup_inputs() dict insertion order, depth-first) ----
  int ix = 0;
  const float*     x   = (const float*)d_in[ix++];       // 0
  const long long* ei  = (const long long*)d_in[ix++];   // 1 (int64 [2,E])
  const float*     img = (const float*)d_in[ix++];       // 2
  const float* embW = (const float*)d_in[ix++];
  const float* embB = (const float*)d_in[ix++];
  struct GatP { const float *W, *as, *ad, *b, *g, *beta; } gat[3];
  for (int i = 0; i < 3; ++i) {
    gat[i].W    = (const float*)d_in[ix++];
    gat[i].as   = (const float*)d_in[ix++];
    gat[i].ad   = (const float*)d_in[ix++];
    gat[i].b    = (const float*)d_in[ix++];
    gat[i].g    = (const float*)d_in[ix++];
    gat[i].beta = (const float*)d_in[ix++];
  }
  struct CnnP { const float *w, *b, *g, *beta; } cnn[4];
  for (int i = 0; i < 4; ++i) {
    cnn[i].w    = (const float*)d_in[ix++];
    cnn[i].b    = (const float*)d_in[ix++];
    cnn[i].g    = (const float*)d_in[ix++];
    cnn[i].beta = (const float*)d_in[ix++];
  }
  const float* projW = (const float*)d_in[ix++];
  const float* projB = (const float*)d_in[ix++];
  const float* Wq = (const float*)d_in[ix++]; const float* bq = (const float*)d_in[ix++];
  const float* Wk = (const float*)d_in[ix++]; const float* bk = (const float*)d_in[ix++];
  const float* Wv = (const float*)d_in[ix++]; const float* bv = (const float*)d_in[ix++];
  const float* Wo = (const float*)d_in[ix++]; const float* bo = (const float*)d_in[ix++];
  const float* d0W = (const float*)d_in[ix++]; const float* d0b = (const float*)d_in[ix++];
  const float* d1W = (const float*)d_in[ix++]; const float* d1b = (const float*)d_in[ix++];
  const float* d2W = (const float*)d_in[ix++]; const float* d2b = (const float*)d_in[ix++];

  // ---- workspace carve ----
  char* p = (char*)d_ws;
  auto alloc = [&](size_t bytes) -> char* {
    char* r = p; p += (bytes + 255) & ~(size_t)255; return r;
  };
  float* bufA = (float*)alloc((size_t)N * 1024 * 4);
  float* bufB = (float*)alloc((size_t)N * 1024 * 4);
  unsigned short* abf = (unsigned short*)alloc((size_t)N * 1024 * 2);
  unsigned short* wbf = (unsigned short*)alloc((size_t)1024 * 1024 * 2);
  float* ssrc = (float*)alloc((size_t)N * 4 * 4);
  float* sdst = (float*)alloc((size_t)N * 4 * 4);
  float* mm   = (float*)alloc((size_t)N * 4 * 4);
  float* den  = (float*)alloc((size_t)N * 4 * 4);
  float* c0   = (float*)alloc((size_t)64 * 32 * 32 * 4);
  float* c1   = (float*)alloc((size_t)64 * 32 * 32 * 4);
  float* mu   = (float*)alloc(512 * 4);
  float* rv   = (float*)alloc(512 * 4);
  float* yv   = (float*)alloc(256 * 4);
  float* kv   = (float*)alloc(256 * 4);

  const int Eraw = in_sizes[1] / 2;     // 320000
  const int E    = Eraw + N;            // + self loops

  auto gemm = [&](const unsigned short* A, const float* W, const float* bias,
                  float* C, int K, int Nc, int act) {
    k_wt_to_bf16_t<<<cdiv(K * Nc, 256), 256, 0, stream>>>(W, wbf, K, Nc);
    dim3 g(cdiv(N / 32, 8), Nc / 64);   // wave macro-tile = 32 rows x 64 cols
    k_wmma_gemm<<<g, 256, 0, stream>>>(A, wbf, bias, C, N, K, Nc, act);
  };

  // ---- embed ----
  k_embed<<<cdiv(N * 128, 256), 256, 0, stream>>>(x, embW, embB, bufA, N);

  // ---- 3 GAT layers ----
  for (int L = 0; L < 3; ++L) {
    int K = (L == 0) ? 128 : HID;
    k_f32_to_bf16<<<cdiv(N * K, 256), 256, 0, stream>>>(bufA, abf, N * K);
    gemm(abf, gat[L].W, nullptr, bufB, K, HID, 0);                     // xl
    k_scores<<<cdiv(N * 4, 256), 256, 0, stream>>>(bufB, gat[L].as, gat[L].ad,
                                                   ssrc, sdst, N);
    k_fill<<<cdiv(N * 4, 256), 256, 0, stream>>>(mm, -INFINITY, N * 4);
    k_fill<<<cdiv(N * 4, 256), 256, 0, stream>>>(den, 0.f, N * 4);
    k_fill<<<cdiv(N * HID, 256), 256, 0, stream>>>(bufA, 0.f, N * HID);
    k_edge_max<<<cdiv(E, 256), 256, 0, stream>>>(ei, ssrc, sdst, mm, Eraw, E);
    k_edge_den<<<cdiv(E, 256), 256, 0, stream>>>(ei, ssrc, sdst, mm, den, Eraw, E);
    k_edge_agg<<<cdiv(E, 8), 256, 0, stream>>>(ei, ssrc, sdst, mm, den, bufB,
                                               bufA, Eraw, E);
    k_bias_ln_elu<<<cdiv(N, 8), 256, 0, stream>>>(bufA, gat[L].b, gat[L].g,
                                                  gat[L].beta, N);
  }

  // ---- CNN branch (tiny) ----
  k_conv<<<cdiv(64 * 32 * 32, 256), 256, 0, stream>>>(img, cnn[0].w, cnn[0].b, c0, 3, 64, 64, 32);
  k_bn_stats<<<64, 256, 0, stream>>>(c0, mu, rv, 32 * 32);
  k_bn_lrelu<<<cdiv(64 * 1024, 256), 256, 0, stream>>>(c0, mu, rv, cnn[0].g, cnn[0].beta, 1024, 64 * 1024);

  k_conv<<<cdiv(128 * 16 * 16, 256), 256, 0, stream>>>(c0, cnn[1].w, cnn[1].b, c1, 64, 32, 128, 16);
  k_bn_stats<<<128, 256, 0, stream>>>(c1, mu, rv, 16 * 16);
  k_bn_lrelu<<<cdiv(128 * 256, 256), 256, 0, stream>>>(c1, mu, rv, cnn[1].g, cnn[1].beta, 256, 128 * 256);

  k_conv<<<cdiv(256 * 8 * 8, 256), 256, 0, stream>>>(c1, cnn[2].w, cnn[2].b, c0, 128, 16, 256, 8);
  k_bn_stats<<<256, 256, 0, stream>>>(c0, mu, rv, 8 * 8);
  k_bn_lrelu<<<cdiv(256 * 64, 256), 256, 0, stream>>>(c0, mu, rv, cnn[2].g, cnn[2].beta, 64, 256 * 64);

  k_conv<<<cdiv(512 * 4 * 4, 256), 256, 0, stream>>>(c0, cnn[3].w, cnn[3].b, c1, 256, 8, 512, 4);
  k_bn_stats<<<512, 256, 0, stream>>>(c1, mu, rv, 4 * 4);
  k_bn_lrelu<<<cdiv(512 * 16, 256), 256, 0, stream>>>(c1, mu, rv, cnn[3].g, cnn[3].beta, 16, 512 * 16);

  k_proj<<<32, 256, 0, stream>>>(c1, projW, projB, yv);
  k_kv<<<1, 256, 0, stream>>>(yv, Wk, bk, Wv, bv, kv);

  // ---- attention (softmax over len-1 axis == 1 -> q + v) ----
  k_f32_to_bf16<<<cdiv(N * HID, 256), 256, 0, stream>>>(bufA, abf, N * HID);
  gemm(abf, Wq, bq, bufB, HID, 128, 0);                    // q [N,128]
  k_addv<<<cdiv(N * 128, 256), 256, 0, stream>>>(bufB, kv + 128, N);
  k_f32_to_bf16<<<cdiv(N * 128, 256), 256, 0, stream>>>(bufB, abf, N * 128);
  gemm(abf, Wo, bo, bufA, 128, HID, 0);                    // h [N,512]

  // ---- decoder ----
  k_f32_to_bf16<<<cdiv(N * HID, 256), 256, 0, stream>>>(bufA, abf, N * HID);
  gemm(abf, d0W, d0b, bufB, HID, 1024, 1);                 // relu
  k_f32_to_bf16<<<cdiv(N * 1024, 256), 256, 0, stream>>>(bufB, abf, N * 1024);
  gemm(abf, d1W, d1b, bufA, 1024, HID, 1);                 // relu
  k_dec2<<<cdiv(N * 4, 256), 256, 0, stream>>>(bufA, d2W, d2b, (float*)d_out, N);
}
